// FeatExtractionEarlyFusion_28269474742324
// MI455X (gfx1250) — matrix operand order repaired and checked by reading, not compiled, not run on hardware
//
#include <hip/hip_runtime.h>
#include <hip/hip_bf16.h>

// ---------------------------------------------------------------------------
// Problem constants (match reference)
// ---------------------------------------------------------------------------
#define BB      4
#define NN      2048
#define NS      64
#define SP_TOT  (4L*2048*64)     // 524288 grouped positions
#define PT_TOT  (4L*2048)        // 8192 points
#define R2      0.09f
#define EPS_GN  1e-5f

typedef __attribute__((ext_vector_type(16))) __bf16 v16bf;
typedef __attribute__((ext_vector_type(8)))  __bf16 v8bf;
typedef __attribute__((ext_vector_type(8)))  float  v8f;

union BV16 { uint4 q[2]; v16bf v; };
union BV8  { uint4 q;    v8bf  v; };

// ---------------------------------------------------------------------------
// Utility kernels
// ---------------------------------------------------------------------------
__global__ void zero_kernel(float* __restrict__ p, int n) {
    int i = blockIdx.x * blockDim.x + threadIdx.x;
    if (i < n) p[i] = 0.f;
}

// f32 weights (Cout x Cin) -> bf16 (Cout x CinPad), zero padded K
__global__ void pack_w_kernel(const float* __restrict__ w, __bf16* __restrict__ o,
                              int Cout, int Cin, int CinPad) {
    long idx = (long)blockIdx.x * blockDim.x + threadIdx.x;
    if (idx >= (long)Cout * CinPad) return;
    int co = (int)(idx / CinPad), ci = (int)(idx % CinPad);
    o[idx] = (ci < Cin) ? (__bf16)w[co * Cin + ci] : (__bf16)0.f;
}

// ---------------------------------------------------------------------------
// Ball-query grouping + fused 10-ch feature (xyz_center, d, ppf) + dxyz
// One wave32 per point; collects first NS in-radius neighbors in index order.
// ---------------------------------------------------------------------------
__device__ __forceinline__ float angle3(float ax, float ay, float az,
                                        float bx, float by, float bz) {
    float cx = ay * bz - az * by;
    float cy = az * bx - ax * bz;
    float cz = ax * by - ay * bx;
    float cr = sqrtf(cx * cx + cy * cy + cz * cz + 1e-12f);
    float dt = ax * bx + ay * by + az * bz;
    return atan2f(cr, dt);
}

__device__ __forceinline__ void write_feat(__bf16* __restrict__ X1,
                                           float* __restrict__ WN0,
                                           long point, int slot,
                                           float cx, float cy, float cz,
                                           float nx, float ny, float nz,
                                           const float* __restrict__ pj,
                                           const float* __restrict__ qj) {
    float dx = pj[0] - cx, dy = pj[1] - cy, dz = pj[2] - cz;
    float gx = qj[0], gy = qj[1], gz = qj[2];
    float a1 = angle3(nx, ny, nz, dx, dy, dz);
    float a2 = angle3(gx, gy, gz, dx, dy, dz);
    float a3 = angle3(nx, ny, nz, gx, gy, gz);
    float dd = sqrtf(dx * dx + dy * dy + dz * dz + 1e-12f);
    long sp = point * NS + slot;
    float f[10] = {cx, cy, cz, dx, dy, dz, a1, a2, a3, dd};
    BV8 o0, o1, o2, o3;
#pragma unroll
    for (int r = 0; r < 8; ++r) o0.v[r] = (__bf16)f[r];
    o1.v[0] = (__bf16)f[8];
    o1.v[1] = (__bf16)f[9];
#pragma unroll
    for (int r = 2; r < 8; ++r) o1.v[r] = (__bf16)0.f;
#pragma unroll
    for (int r = 0; r < 8; ++r) { o2.v[r] = (__bf16)0.f; o3.v[r] = (__bf16)0.f; }
    uint4* dst = (uint4*)(X1 + sp * 32);
    dst[0] = o0.q; dst[1] = o1.q; dst[2] = o2.q; dst[3] = o3.q;
    *(float4*)(WN0 + sp * 4) = make_float4(dx, dy, dz, 0.f);
}

__global__ __launch_bounds__(128)
void group_fused_kernel(const float* __restrict__ xyz, const float* __restrict__ nrm,
                        __bf16* __restrict__ X1, float* __restrict__ WN0) {
    const int lane = threadIdx.x & 31;
    const long point = (long)blockIdx.x * 4 + (threadIdx.x >> 5);
    const int b = (int)(point >> 11);
    const int i = (int)(point & (NN - 1));
    const float* cb = xyz + ((long)b * NN + i) * 3;
    const float cx = cb[0], cy = cb[1], cz = cb[2];
    const float* nb = nrm + ((long)b * NN + i) * 3;
    const float nx = nb[0], ny = nb[1], nz = nb[2];

    int count = 0;
    int jf = -1;
    for (int base = 0; base < NN; base += 32) {
        int j = base + lane;
        const float* p = xyz + ((long)b * NN + j) * 3;
        float dx = p[0] - cx, dy = p[1] - cy, dz = p[2] - cz;
        float sq = dx * dx + dy * dy + dz * dz;
        bool pred = (sq <= R2);
        unsigned int mask = (unsigned int)__ballot(pred);
        if (jf < 0 && mask) jf = base + __builtin_ctz(mask);
        if (pred) {
            int slot = count + __popc(mask & ((1u << lane) - 1u));
            if (slot < NS)
                write_feat(X1, WN0, point, slot, cx, cy, cz, nx, ny, nz,
                           p, nrm + ((long)b * NN + j) * 3);
        }
        count += __popc(mask);
        if (count >= NS) break;
    }
    if (count < NS && jf >= 0) {  // pad with first neighbor (self always in-radius)
        const float* p = xyz + ((long)b * NN + jf) * 3;
        const float* q = nrm + ((long)b * NN + jf) * 3;
        for (int slot = count + lane; slot < NS; slot += 32)
            write_feat(X1, WN0, point, slot, cx, cy, cz, nx, ny, nz, p, q);
    }
}

// ---------------------------------------------------------------------------
// WMMA bf16 GEMM:  Y[sp][co] = sum_ci Wp[co][ci]*X[sp][ci] + bias[co]
// X: [NSP][CIN] bf16 (dense rows). Wp: [COUT][CIN] bf16. CIN % 32 == 0.
// One wave per (16-sp x MCO*16-co) macro tile: one streamed B load feeds MCO
// v_wmma_f32_16x16x32_bf16 issues (register-level B reuse).
// ---------------------------------------------------------------------------
template <int MCO>
__global__ __launch_bounds__(128)
void gemm_bf16_kernel(const __bf16* __restrict__ X, const __bf16* __restrict__ Wp,
                      const float* __restrict__ bias, void* __restrict__ Yout,
                      int CIN, int COUT, long NSP, int yIsF32) {
    const int lane = threadIdx.x & 31;
    const long wave = (long)blockIdx.x * 4 + (threadIdx.x >> 5);
    const int nChunk = (COUT >> 4) / MCO;
    const long tasks = (NSP >> 4) * nChunk;
    if (wave >= tasks) return;
    const int chunk = (int)(wave % nChunk);
    const long spT = wave / nChunk;
    const int half = lane >> 4;
    const int l16 = lane & 15;
    const long spCol = spT * 16 + l16;                 // this lane's N column (sp)
    const int coRow0 = chunk * MCO * 16 + l16;         // first co tile's M row

    const int KT = CIN >> 5;
    const __bf16* brow = X + spCol * CIN + half * 16;  // K = half*16..+15
    const __bf16* arow0 = Wp + (long)coRow0 * CIN + half * 16;

    v8f acc[MCO];
#pragma unroll
    for (int m = 0; m < MCO; ++m)
        acc[m] = (v8f){0.f, 0.f, 0.f, 0.f, 0.f, 0.f, 0.f, 0.f};

    for (int kt = 0; kt < KT; ++kt) {
        BV16 bmat;
        const __bf16* bp = brow + kt * 32;
        bmat.q[0] = *(const uint4*)(bp);
        bmat.q[1] = *(const uint4*)(bp + 8);
        __builtin_prefetch(bp + 32, 0, 3);   // near-cache prefetch of next K tile
#pragma unroll
        for (int m = 0; m < MCO; ++m) {
            BV16 a;
            const __bf16* ap = arow0 + (long)m * 16 * CIN + kt * 32;
            a.q[0] = *(const uint4*)(ap);
            a.q[1] = *(const uint4*)(ap + 8);
            acc[m] = __builtin_amdgcn_wmma_f32_16x16x32_bf16(false, a.v, false, bmat.v,
                                                            (short)0, acc[m], false, false);
        }
    }

#pragma unroll
    for (int m = 0; m < MCO; ++m) {
        const int coBase = chunk * MCO * 16 + m * 16 + half * 8;
        float tv[8];
#pragma unroll
        for (int r = 0; r < 8; ++r) tv[r] = acc[m][r] + bias[coBase + r];
        if (yIsF32) {
            float* Yf = (float*)Yout;
            float* d = Yf + spCol * COUT + coBase;
            *(float4*)(d)     = make_float4(tv[0], tv[1], tv[2], tv[3]);
            *(float4*)(d + 4) = make_float4(tv[4], tv[5], tv[6], tv[7]);
        } else {
            __bf16* Yb = (__bf16*)Yout;
            BV8 o;
#pragma unroll
            for (int r = 0; r < 8; ++r) o.v[r] = (__bf16)tv[r];
            *(uint4*)(Yb + spCol * COUT + coBase) = o.q;
        }
    }
}

// ---------------------------------------------------------------------------
// GroupNorm statistics: per (batch, group) sum & sumsq with atomics.
// G is compile-time so accumulators live in registers.
// ---------------------------------------------------------------------------
template <typename T, int G>
__global__ __launch_bounds__(256)
void stats_kernel(const T* __restrict__ X, float* __restrict__ stats,
                  int C, int strideC, long rowsPerBatch, int cpg) {
    __shared__ float ss[G], sq[G];
    int tid = threadIdx.x;
    if (tid < G) { ss[tid] = 0.f; sq[tid] = 0.f; }
    __syncthreads();
    int b = blockIdx.y;
    float acc[G], ac2[G];
#pragma unroll
    for (int g = 0; g < G; ++g) { acc[g] = 0.f; ac2[g] = 0.f; }
    for (long row = (long)blockIdx.x * blockDim.x + tid; row < rowsPerBatch;
         row += (long)gridDim.x * blockDim.x) {
        const T* p = X + ((long)b * rowsPerBatch + row) * strideC;
        int cbase = 0;
#pragma unroll
        for (int g = 0; g < G; ++g) {
            float a = 0.f, a2 = 0.f;
            for (int k = 0; k < cpg; ++k) {
                float v = (float)p[cbase + k];
                a += v; a2 += v * v;
            }
            acc[g] += a; ac2[g] += a2; cbase += cpg;
        }
    }
#pragma unroll
    for (int g = 0; g < G; ++g) { atomicAdd(&ss[g], acc[g]); atomicAdd(&sq[g], ac2[g]); }
    __syncthreads();
    if (tid < G) {
        atomicAdd(&stats[(b * G + tid) * 2 + 0], ss[tid]);
        atomicAdd(&stats[(b * G + tid) * 2 + 1], sq[tid]);
    }
}

// GN + ReLU, in place
template <typename T>
__global__ __launch_bounds__(256)
void act_kernel(T* __restrict__ X, const float* __restrict__ stats,
                const float* __restrict__ gamma, const float* __restrict__ beta,
                int C, int strideC, long rowsPerBatch, int G, int cpg, float invCnt) {
    long row = (long)blockIdx.x * blockDim.x + threadIdx.x;
    long totalRows = (long)BB * rowsPerBatch;
    if (row >= totalRows) return;
    int b = (int)(row / rowsPerBatch);
    T* p = X + row * strideC;
    const float* st = stats + (long)b * G * 2;
    int g = 0, k = 0;
    float mean = 0.f, sc = 0.f;
    for (int c = 0; c < C; ++c) {
        if (k == 0) {
            float s0 = st[g * 2], s1 = st[g * 2 + 1];
            mean = s0 * invCnt;
            float var = s1 * invCnt - mean * mean;
            sc = rsqrtf(var + EPS_GN);
        }
        float v = ((float)p[c] - mean) * sc * gamma[c] + beta[c];
        p[c] = (T)(v > 0.f ? v : 0.f);
        if (++k == cpg) { k = 0; ++g; }
    }
}

// ---------------------------------------------------------------------------
// WeightNet small conv (f32, Cin/Cout <= 8)
// ---------------------------------------------------------------------------
template <int CIN, int COUT>
__global__ __launch_bounds__(256)
void wn_conv_kernel(const float* __restrict__ X, int strideIn,
                    const float* __restrict__ W, const float* __restrict__ bias,
                    float* __restrict__ Y, int strideOut, long nsp) {
    __shared__ float sw[CIN * COUT];
    __shared__ float sb[COUT];
    int tid = threadIdx.x;
    if (tid < CIN * COUT) sw[tid] = W[tid];
    if (tid < COUT) sb[tid] = bias[tid];
    __syncthreads();
    long sp = (long)blockIdx.x * blockDim.x + tid;
    if (sp >= nsp) return;
    float xin[CIN];
#pragma unroll
    for (int ci = 0; ci < CIN; ++ci) xin[ci] = X[sp * strideIn + ci];
#pragma unroll
    for (int co = 0; co < COUT; ++co) {
        float a = sb[co];
#pragma unroll
        for (int ci = 0; ci < CIN; ++ci) a += sw[co * CIN + ci] * xin[ci];
        Y[sp * strideOut + co] = a;
    }
}

// ---------------------------------------------------------------------------
// pooled[p][c*8+w] = sum_s X3[(p*64+s)][c] * wts[(p*64+s)][w]
// ---------------------------------------------------------------------------
__global__ __launch_bounds__(192)
void pool_kernel(const __bf16* __restrict__ X3, const float* __restrict__ WTS,
                 __bf16* __restrict__ pooled) {
    __shared__ float sw[NS * 8];
    long p = blockIdx.x;
    int tid = threadIdx.x;
    for (int idx = tid; idx < NS * 8; idx += 192) sw[idx] = WTS[p * NS * 8 + idx];
    __syncthreads();
    int c = tid;  // 0..191
    float acc[8];
#pragma unroll
    for (int w = 0; w < 8; ++w) acc[w] = 0.f;
    const __bf16* xb = X3 + p * NS * 192 + c;
    for (int s = 0; s < NS; ++s) {
        float xv = (float)xb[(long)s * 192];
#pragma unroll
        for (int w = 0; w < 8; ++w) acc[w] += xv * sw[s * 8 + w];
    }
    BV8 o;
#pragma unroll
    for (int w = 0; w < 8; ++w) o.v[w] = (__bf16)acc[w];
    *(uint4*)(pooled + p * 1536 + c * 8) = o.q;
}

// Final L2-normalize over 96 channels, one wave per point
__global__ __launch_bounds__(128)
void normalize_kernel(const float* __restrict__ Y, float* __restrict__ out) {
    int lane = threadIdx.x & 31;
    long p = (long)blockIdx.x * 4 + (threadIdx.x >> 5);
    const float* yb = Y + p * 96;
    float v0 = yb[lane], v1 = yb[lane + 32], v2 = yb[lane + 64];
    float ss = v0 * v0 + v1 * v1 + v2 * v2;
    for (int off = 16; off > 0; off >>= 1) ss += __shfl_xor(ss, off, 32);
    float s = 1.0f / sqrtf(ss);
    float* ob = out + p * 96;
    ob[lane] = v0 * s; ob[lane + 32] = v1 * s; ob[lane + 64] = v2 * s;
}

// ---------------------------------------------------------------------------
// Workspace layout (bytes, 256-aligned)
// ---------------------------------------------------------------------------
static constexpr size_t OFF_X1   = 0;                                   // [SP][32] bf16
static constexpr size_t OFF_P0   = OFF_X1   + SP_TOT * 32 * 2;          // [SP][<=192] bf16
static constexpr size_t OFF_P1   = OFF_P0   + SP_TOT * 192 * 2;
static constexpr size_t OFF_WN0  = OFF_P1   + SP_TOT * 192 * 2;         // [SP][4] f32
static constexpr size_t OFF_WN1  = OFF_WN0  + SP_TOT * 4 * 4;           // [SP][8] f32
static constexpr size_t OFF_WN2  = OFF_WN1  + SP_TOT * 8 * 4;
static constexpr size_t OFF_POOL = OFF_WN2  + SP_TOT * 8 * 4;           // [PT][1536] bf16
static constexpr size_t OFF_Y4   = OFF_POOL + PT_TOT * 1536 * 2;        // [PT][192] bf16
static constexpr size_t OFF_Y5   = OFF_Y4   + PT_TOT * 192 * 2;
static constexpr size_t OFF_Y6   = OFF_Y5   + PT_TOT * 192 * 2;         // [PT][96] bf16
static constexpr size_t OFF_Y7   = OFF_Y6   + PT_TOT * 96 * 2;          // [PT][96] f32
static constexpr size_t OFF_WPP0 = OFF_Y7   + PT_TOT * 96 * 4;
static constexpr size_t OFF_WPP1 = OFF_WPP0 + 96 * 32 * 2;
static constexpr size_t OFF_WPP2 = OFF_WPP1 + 96 * 96 * 2;
static constexpr size_t OFF_WLIN = OFF_WPP2 + 192 * 96 * 2;
static constexpr size_t OFF_WPO1 = OFF_WLIN + 192L * 1536 * 2;
static constexpr size_t OFF_WPO2 = OFF_WPO1 + 192 * 192 * 2;
static constexpr size_t OFF_WPO3 = OFF_WPO2 + 96 * 192 * 2;
static constexpr size_t OFF_STAT = OFF_WPO3 + 96 * 96 * 2;              // 9 slots x 64 f32

extern "C" void kernel_launch(void* const* d_in, const int* in_sizes, int n_in,
                              void* d_out, int out_size, void* d_ws, size_t ws_size,
                              hipStream_t stream) {
    (void)in_sizes; (void)n_in; (void)out_size; (void)ws_size;
    // Input flattening: setup_inputs() insertion order: xyz, normals, then
    // params in insertion order {pp[3x(w,b,g,be)], wn[3x(w,b,g,be)], lin(w,b),
    // bnl(g,be), po[(w,b,g,be),(w,b,g,be),(w,b)]}.
    const float* xyz  = (const float*)d_in[0];
    const float* nrm  = (const float*)d_in[1];
    const float* pp0w = (const float*)d_in[2],  *pp0b = (const float*)d_in[3];
    const float* pp0g = (const float*)d_in[4],  *pp0e = (const float*)d_in[5];
    const float* pp1w = (const float*)d_in[6],  *pp1b = (const float*)d_in[7];
    const float* pp1g = (const float*)d_in[8],  *pp1e = (const float*)d_in[9];
    const float* pp2w = (const float*)d_in[10], *pp2b = (const float*)d_in[11];
    const float* pp2g = (const float*)d_in[12], *pp2e = (const float*)d_in[13];
    const float* wn0w = (const float*)d_in[14], *wn0b = (const float*)d_in[15];
    const float* wn0g = (const float*)d_in[16], *wn0e = (const float*)d_in[17];
    const float* wn1w = (const float*)d_in[18], *wn1b = (const float*)d_in[19];
    const float* wn1g = (const float*)d_in[20], *wn1e = (const float*)d_in[21];
    const float* wn2w = (const float*)d_in[22], *wn2b = (const float*)d_in[23];
    const float* wn2g = (const float*)d_in[24], *wn2e = (const float*)d_in[25];
    const float* linw = (const float*)d_in[26], *linb = (const float*)d_in[27];
    const float* bnlg = (const float*)d_in[28], *bnle = (const float*)d_in[29];
    const float* po1w = (const float*)d_in[30], *po1b = (const float*)d_in[31];
    const float* po1g = (const float*)d_in[32], *po1e = (const float*)d_in[33];
    const float* po2w = (const float*)d_in[34], *po2b = (const float*)d_in[35];
    const float* po2g = (const float*)d_in[36], *po2e = (const float*)d_in[37];
    const float* po3w = (const float*)d_in[38], *po3b = (const float*)d_in[39];

    char* ws = (char*)d_ws;
    __bf16* X1   = (__bf16*)(ws + OFF_X1);
    __bf16* P0   = (__bf16*)(ws + OFF_P0);
    __bf16* P1   = (__bf16*)(ws + OFF_P1);
    float*  WN0  = (float*)(ws + OFF_WN0);
    float*  WN1  = (float*)(ws + OFF_WN1);
    float*  WN2  = (float*)(ws + OFF_WN2);
    __bf16* POOL = (__bf16*)(ws + OFF_POOL);
    __bf16* Y4   = (__bf16*)(ws + OFF_Y4);
    __bf16* Y5   = (__bf16*)(ws + OFF_Y5);
    __bf16* Y6   = (__bf16*)(ws + OFF_Y6);
    float*  Y7   = (float*)(ws + OFF_Y7);
    __bf16* Wpp0 = (__bf16*)(ws + OFF_WPP0);
    __bf16* Wpp1 = (__bf16*)(ws + OFF_WPP1);
    __bf16* Wpp2 = (__bf16*)(ws + OFF_WPP2);
    __bf16* Wlin = (__bf16*)(ws + OFF_WLIN);
    __bf16* Wpo1 = (__bf16*)(ws + OFF_WPO1);
    __bf16* Wpo2 = (__bf16*)(ws + OFF_WPO2);
    __bf16* Wpo3 = (__bf16*)(ws + OFF_WPO3);
    float*  ST   = (float*)(ws + OFF_STAT);   // 9 slots of 64 floats

    const long SPR = NN * (long)NS;  // rows per batch, grouped space (131072)
    const long PTR = NN;             // rows per batch, point space (2048)

    zero_kernel<<<3, 256, 0, stream>>>(ST, 9 * 64);

    auto packN = [&](const float* w, __bf16* o, int co, int ci, int cip) {
        long n = (long)co * cip;
        pack_w_kernel<<<(int)((n + 255) / 256), 256, 0, stream>>>(w, o, co, ci, cip);
    };
    packN(pp0w, Wpp0, 96, 10, 32);
    packN(pp1w, Wpp1, 96, 96, 96);
    packN(pp2w, Wpp2, 192, 96, 96);
    packN(linw, Wlin, 192, 1536, 1536);
    packN(po1w, Wpo1, 192, 192, 192);
    packN(po2w, Wpo2, 96, 192, 192);
    packN(po3w, Wpo3, 96, 96, 96);

    group_fused_kernel<<<(int)(PT_TOT / 4), 128, 0, stream>>>(xyz, nrm, X1, WN0);

    // MCO = co-tiles per wave: 3 for COUT=96, 4 for COUT=192
    auto gemm = [&](const __bf16* X, const __bf16* Wp, const float* bias, void* Y,
                    int CIN, int COUT, long NSP, int f32o) {
        if (COUT == 96) {
            long tasks = (NSP >> 4) * ((COUT >> 4) / 3);
            gemm_bf16_kernel<3><<<(int)((tasks + 3) >> 2), 128, 0, stream>>>(
                X, Wp, bias, Y, CIN, COUT, NSP, f32o);
        } else {
            long tasks = (NSP >> 4) * ((COUT >> 4) / 4);
            gemm_bf16_kernel<4><<<(int)((tasks + 3) >> 2), 128, 0, stream>>>(
                X, Wp, bias, Y, CIN, COUT, NSP, f32o);
        }
    };
    auto actB = [&](__bf16* X, float* st, const float* g, const float* e,
                    int C, long rpb, int G, int cpg) {
        long rows = BB * rpb;
        act_kernel<__bf16><<<(int)((rows + 255) / 256), 256, 0, stream>>>(
            X, st, g, e, C, C, rpb, G, cpg, 1.0f / (float)(rpb * cpg));
    };
    auto actF = [&](float* X, float* st, const float* g, const float* e,
                    int C, int strd, long rpb, int G, int cpg) {
        long rows = BB * rpb;
        act_kernel<float><<<(int)((rows + 255) / 256), 256, 0, stream>>>(
            X, st, g, e, C, strd, rpb, G, cpg, 1.0f / (float)(rpb * cpg));
    };

    // ---- prepool: 10->96 -> 96->96 -> 96->192 (GN8 + ReLU between) ----
    gemm(X1, Wpp0, pp0b, P0, 32, 96, SP_TOT, 0);
    stats_kernel<__bf16, 8><<<dim3(64, BB), 256, 0, stream>>>(P0, ST + 0 * 64, 96, 96, SPR, 12);
    actB(P0, ST + 0 * 64, pp0g, pp0e, 96, SPR, 8, 12);

    gemm(P0, Wpp1, pp1b, P1, 96, 96, SP_TOT, 0);
    stats_kernel<__bf16, 8><<<dim3(64, BB), 256, 0, stream>>>(P1, ST + 1 * 64, 96, 96, SPR, 12);
    actB(P1, ST + 1 * 64, pp1g, pp1e, 96, SPR, 8, 12);

    gemm(P1, Wpp2, pp2b, P0, 96, 192, SP_TOT, 0);
    stats_kernel<__bf16, 8><<<dim3(64, BB), 256, 0, stream>>>(P0, ST + 2 * 64, 192, 192, SPR, 24);
    actB(P0, ST + 2 * 64, pp2g, pp2e, 192, SPR, 8, 24);

    // ---- WeightNet: 3->4 -> 4->4 -> 4->8 (GN4 + ReLU) ----
    int wng = (int)((SP_TOT + 255) / 256);
    wn_conv_kernel<3, 4><<<wng, 256, 0, stream>>>(WN0, 4, wn0w, wn0b, WN1, 8, SP_TOT);
    stats_kernel<float, 4><<<dim3(64, BB), 256, 0, stream>>>(WN1, ST + 3 * 64, 4, 8, SPR, 1);
    actF(WN1, ST + 3 * 64, wn0g, wn0e, 4, 8, SPR, 4, 1);

    wn_conv_kernel<4, 4><<<wng, 256, 0, stream>>>(WN1, 8, wn1w, wn1b, WN2, 8, SP_TOT);
    stats_kernel<float, 4><<<dim3(64, BB), 256, 0, stream>>>(WN2, ST + 4 * 64, 4, 8, SPR, 1);
    actF(WN2, ST + 4 * 64, wn1g, wn1e, 4, 8, SPR, 4, 1);

    wn_conv_kernel<4, 8><<<wng, 256, 0, stream>>>(WN2, 8, wn2w, wn2b, WN1, 8, SP_TOT);
    stats_kernel<float, 4><<<dim3(64, BB), 256, 0, stream>>>(WN1, ST + 5 * 64, 8, 8, SPR, 2);
    actF(WN1, ST + 5 * 64, wn2g, wn2e, 8, 8, SPR, 4, 2);

    // ---- weighted pool + linear 1536->192 (+ GN8 + ReLU) ----
    pool_kernel<<<(int)PT_TOT, 192, 0, stream>>>(P0, WN1, POOL);
    gemm(POOL, Wlin, linb, Y4, 1536, 192, PT_TOT, 0);
    stats_kernel<__bf16, 8><<<dim3(64, BB), 256, 0, stream>>>(Y4, ST + 6 * 64, 192, 192, PTR, 24);
    actB(Y4, ST + 6 * 64, bnlg, bnle, 192, PTR, 8, 24);

    // ---- postpool ----
    gemm(Y4, Wpo1, po1b, Y5, 192, 192, PT_TOT, 0);
    stats_kernel<__bf16, 8><<<dim3(64, BB), 256, 0, stream>>>(Y5, ST + 7 * 64, 192, 192, PTR, 24);
    actB(Y5, ST + 7 * 64, po1g, po1e, 192, PTR, 8, 24);

    gemm(Y5, Wpo2, po2b, Y6, 192, 96, PT_TOT, 0);
    stats_kernel<__bf16, 8><<<dim3(64, BB), 256, 0, stream>>>(Y6, ST + 8 * 64, 96, 96, PTR, 12);
    actB(Y6, ST + 8 * 64, po2g, po2e, 96, PTR, 8, 12);

    gemm(Y6, Wpo3, po3b, Y7, 96, 96, PT_TOT, 1);  // f32 out

    normalize_kernel<<<(int)(PT_TOT / 4), 128, 0, stream>>>(Y7, (float*)d_out);
}
